// CSVT_16037407883468
// MI455X (gfx1250) — matrix operand order repaired and checked by reading, not compiled, not run on hardware
//
#include <hip/hip_runtime.h>
#include <cstdint>
#include <cstddef>

#define NPTS 131072
#define CCH  256
#define TTOK 16
#define BSEG 8
#define WIDE 576   // 256 (xq) + 256 (xp) + 16 (xk) padded to 9*64
#define EPSV 1e-5f

typedef __bf16 bf16_t;
typedef bf16_t bf16x16 __attribute__((ext_vector_type(16)));
typedef bf16_t bf16x4  __attribute__((ext_vector_type(4)));
typedef float  f32x8   __attribute__((ext_vector_type(8)));

#define NCF_CAT (WIDE / 16)   // 36 col fragments for the fused weight
#define NCF_TR  (256 / 16)    // 16 col fragments for Wtrans
#define AF_STRIDE 24          // padded lane record (16 used), 48B, 16B-multiple

// ---------------------------------------------------------------------------
// bf16 WMMA GEMM:  Y[M, ncols] = A[M, 256](f32) * Wf(frag-swizzled bf16)
// Block: 256 threads = 8 waves; tile 128 rows x 64 cols; K = 8 x 32, unrolled.
// B fragments are read as contiguous 32B records straight from global (L2-hot).
// A is staged through LDS in fragment order: gather = 2 x ds_load_b128/lane.
// ---------------------------------------------------------------------------
__launch_bounds__(256)
__global__ void k_gemm_bf16(const float* __restrict__ A, int lda,
                            const bf16_t* __restrict__ Wf,
                            float* __restrict__ Y, int ldy) {
    __shared__ bf16_t Af[8 * 32 * AF_STRIDE];   // 8 row-subtiles x 32 lanes

    const int tid  = threadIdx.x;
    const int wv   = tid >> 5;          // wave id 0..7 -> row sub-tile
    const int ln   = tid & 31;          // lane id
    const int row0 = blockIdx.x * 128;
    const int col0 = blockIdx.y * 64;
    const int cf0  = blockIdx.y * 4;    // base col-fragment (16 cols each)

    f32x8 acc[4];
#pragma unroll
    for (int i = 0; i < 4; ++i)
#pragma unroll
        for (int j = 0; j < 8; ++j) acc[i][j] = 0.0f;

#pragma unroll
    for (int kb = 0; kb < 8; ++kb) {
        const int k0 = kb * 32;
        // ---- stage A tile (f32 -> bf16) into FRAGMENT order ----
        // fragment record for (subtile, lane): 16 bf16, lane/elem in contiguous order k = 16h + s
#pragma unroll
        for (int i = 0; i < 4; ++i) {
            int idx = tid + i * 256;            // 1024 float4 slots
            int r   = idx >> 3;                 // row 0..127
            int kq  = (idx & 7) << 2;           // k offset 0,4,..,28
            const float4 v = *reinterpret_cast<const float4*>(
                A + (size_t)(row0 + r) * lda + k0 + kq);
            int chunk = kq >> 3;                            // 8-aligned k chunk
            int lane  = (r & 15) + ((kq & 16) ? 16 : 0);    // contiguous: k = 16h + s
            int elem  = kq & 15;                            // elem 0..12
            (void)chunk;
            int rec   = ((r >> 4) * 32 + lane) * AF_STRIDE + elem;
            bf16x4 t;
            t[0] = (bf16_t)v.x; t[1] = (bf16_t)v.y;
            t[2] = (bf16_t)v.z; t[3] = (bf16_t)v.w;
            *reinterpret_cast<bf16x4*>(&Af[rec]) = t;       // 8B packed store
        }
        if (k0 + 32 < 256)
            __builtin_prefetch(A + (size_t)(row0 + (tid >> 1)) * lda + k0 + 32, 0, 1);
        __syncthreads();

        // ---- A fragment: one 32B contiguous LDS record per lane ----
        bf16x16 afrag;
        __builtin_memcpy(&afrag, &Af[(wv * 32 + ln) * AF_STRIDE], 32);

        // ---- 4 col fragments: contiguous 32B global records, WMMA accumulate ----
#pragma unroll
        for (int cf = 0; cf < 4; ++cf) {
            bf16x16 bfrag;
            const bf16_t* bp = Wf + (((size_t)(cf0 + cf) * 8 + kb) * 32 + ln) * 16;
            __builtin_memcpy(&bfrag, bp, 32);
            acc[cf] = __builtin_amdgcn_wmma_f32_16x16x32_bf16(
                false, afrag, false, bfrag, (short)0, acc[cf], false, false);
        }
        __syncthreads();
    }

    // ---- store C/D: lanes 0-15 rows r, lanes 16-31 rows r+8 ----
    const int rbase = row0 + wv * 16 + ((ln & 16) ? 8 : 0);
    const int cbase = col0 + (ln & 15);
#pragma unroll
    for (int cf = 0; cf < 4; ++cf)
#pragma unroll
        for (int r = 0; r < 8; ++r)
            Y[(size_t)(rbase + r) * ldy + cbase + cf * 16] = acc[cf][r];
}

// ---------------------------------------------------------------------------
// Weight packing into WMMA-fragment-ready global layout:
//   Wf[cf][kb][lane][elem] ; c = cf*16 + (lane&15) ; k = kb*32 + elem + (lane&16?16:0)
// WcatF = [Wq | Wp | Wk | 0-pad] (36 cf), WtbF = Wtrans (16 cf)
// ---------------------------------------------------------------------------
__global__ void k_convert(const float* __restrict__ Wq, const float* __restrict__ Wk,
                          const float* __restrict__ Wp, const float* __restrict__ Wtr,
                          bf16_t* __restrict__ WcatF, bf16_t* __restrict__ WtbF) {
    int idx = blockIdx.x * 256 + threadIdx.x;
    if (idx < NCF_CAT * 8 * 512) {
        int e  = idx & 15;
        int ln = (idx >> 4) & 31;
        int kb = (idx >> 9) & 7;
        int cf = idx >> 12;
        int c  = cf * 16 + (ln & 15);
        int k  = kb * 32 + e + ((ln & 16) ? 16 : 0);
        float v = 0.0f;
        if (c < 256)      v = Wq[k * 256 + c];
        else if (c < 512) v = Wp[k * 256 + (c - 256)];
        else if (c < 528) v = Wk[k * 16 + (c - 512)];
        WcatF[idx] = (bf16_t)v;
    } else {
        int j = idx - NCF_CAT * 8 * 512;
        if (j < NCF_TR * 8 * 512) {
            int e  = j & 15;
            int ln = (j >> 4) & 31;
            int kb = (j >> 9) & 7;
            int cf = j >> 12;
            int c  = cf * 16 + (ln & 15);
            int k  = kb * 32 + e + ((ln & 16) ? 16 : 0);
            WtbF[j] = (bf16_t)Wtr[k * 256 + c];
        }
    }
}

// Segment boundaries via binary search (batch_ids sorted): start[b], start[B]=N
__global__ void k_bounds(const int* __restrict__ bid, int* __restrict__ start) {
    int t = threadIdx.x;
    if (t <= BSEG) {
        int lo = 0, hi = NPTS;
        while (lo < hi) { int mid = (lo + hi) >> 1; if (bid[mid] < t) lo = mid + 1; else hi = mid; }
        start[t] = lo;
    }
}

// Per-(b,t) segment max of xk (cols 512..527 of Ycat)
__global__ void k_segmax(const float* __restrict__ Ycat, const int* __restrict__ start,
                         float* __restrict__ m) {
    __shared__ float sd[256];
    int b = blockIdx.x >> 4, t = blockIdx.x & 15;
    int s0 = start[b], s1 = start[b + 1];
    float mx = -3.0e38f;
    for (int n = s0 + threadIdx.x; n < s1; n += 256)
        mx = fmaxf(mx, Ycat[(size_t)n * WIDE + 512 + t]);
    sd[threadIdx.x] = mx; __syncthreads();
    for (int s = 128; s > 0; s >>= 1) {
        if (threadIdx.x < s) sd[threadIdx.x] = fmaxf(sd[threadIdx.x], sd[threadIdx.x + s]);
        __syncthreads();
    }
    if (threadIdx.x == 0) m[b * 16 + t] = sd[0];
}

// Per-(b,t) segment sum of exp(xk - m)
__global__ void k_segsum(const float* __restrict__ Ycat, const int* __restrict__ start,
                         const float* __restrict__ m, float* __restrict__ z) {
    __shared__ float sd[256];
    int b = blockIdx.x >> 4, t = blockIdx.x & 15;
    int s0 = start[b], s1 = start[b + 1];
    float mv = m[b * 16 + t];
    float s = 0.0f;
    for (int n = s0 + threadIdx.x; n < s1; n += 256)
        s += expf(Ycat[(size_t)n * WIDE + 512 + t] - mv);
    sd[threadIdx.x] = s; __syncthreads();
    for (int st = 128; st > 0; st >>= 1) {
        if (threadIdx.x < st) sd[threadIdx.x] += sd[threadIdx.x + st];
        __syncthreads();
    }
    if (threadIdx.x == 0) z[b * 16 + t] = sd[0];
}

// tokens partials: block (b, cc, chunk); thread (c_local, t); no atomics
__global__ void k_tokens(const float* __restrict__ Ycat, const int* __restrict__ start,
                         const float* __restrict__ m, const float* __restrict__ z,
                         float* __restrict__ part) {
    int b = blockIdx.x, cc = blockIdx.y, ch = blockIdx.z;   // 8 x 16 x 32
    int t = threadIdx.x & 15, cl = threadIdx.x >> 4;
    int c = cc * 16 + cl;
    int s0 = start[b], s1 = start[b + 1];
    float mv = m[b * 16 + t];
    float zv = z[b * 16 + t];
    float zinv = (zv > 0.0f) ? 1.0f / zv : 0.0f;
    float acc = 0.0f;
    for (int n = s0 + ch; n < s1; n += 32) {
        float kv = Ycat[(size_t)n * WIDE + 512 + t];
        float xv = Ycat[(size_t)n * WIDE + c];
        acc += expf(kv - mv) * zinv * xv;
    }
    part[(size_t)ch * (BSEG * CCH * TTOK) + ((b * CCH + c) * TTOK + t)] = acc;
}

__global__ void k_tokred(const float* __restrict__ part, float* __restrict__ tokens) {
    int idx = blockIdx.x * 256 + threadIdx.x;   // < 32768
    float s = 0.0f;
    for (int ch = 0; ch < 32; ++ch) s += part[(size_t)ch * 32768 + idx];
    tokens[idx] = s;
}

// ---------------------------------------------------------------------------
// Token self-attention, one block per cloud, all intermediates in LDS.
// ---------------------------------------------------------------------------
__launch_bounds__(256)
__global__ void k_tokattn(const float* __restrict__ tokens,
                          const float* __restrict__ Wv, const float* __restrict__ Wke,
                          const float* __restrict__ Wqe, const float* __restrict__ Wemb,
                          const float* __restrict__ Wt, float* __restrict__ TP) {
    __shared__ float tok[CCH * TTOK];
    __shared__ float bufA[CCH * TTOK];
    __shared__ float bufB[CCH * TTOK];
    __shared__ float att[TTOK * TTOK];
    int b = blockIdx.x, tid = threadIdx.x;

    for (int i = 0; i < 16; ++i) tok[tid + i * 256] = tokens[b * 4096 + tid + i * 256];
    __syncthreads();

    // keys -> bufA, qrs -> bufB  (out[o][t] = sum_c W[o][c]*tok[c][t])
    {
        float accA[16], accB[16];
        for (int t = 0; t < 16; ++t) { accA[t] = 0.0f; accB[t] = 0.0f; }
        for (int c = 0; c < 256; ++c) {
            float wk = Wke[tid * 256 + c], wq = Wqe[tid * 256 + c];
            for (int t = 0; t < 16; ++t) { float tv = tok[c * 16 + t]; accA[t] += wk * tv; accB[t] += wq * tv; }
        }
        for (int t = 0; t < 16; ++t) { bufA[tid * 16 + t] = accA[t]; bufB[tid * 16 + t] = accB[t]; }
    }
    __syncthreads();
    // att[t][s] = sum_c keys[c][t]*qrs[c][s]
    {
        int t = tid >> 4, s = tid & 15;
        float a = 0.0f;
        for (int c = 0; c < 256; ++c) a += bufA[c * 16 + t] * bufB[c * 16 + s];
        att[t * 16 + s] = a;
    }
    __syncthreads();
    // softmax over s
    {
        int t = tid >> 4, s = tid & 15;
        float mx = -3.0e38f;
        for (int j = 0; j < 16; ++j) mx = fmaxf(mx, att[t * 16 + j]);
        float e = expf(att[t * 16 + s] - mx);
        __syncthreads();
        att[t * 16 + s] = e;
        __syncthreads();
        float sum = 0.0f;
        for (int j = 0; j < 16; ++j) sum += att[t * 16 + j];
        float w = e / sum;
        __syncthreads();
        att[t * 16 + s] = w;
    }
    __syncthreads();
    // vals -> bufA (keys no longer needed)
    {
        float acc[16];
        for (int t = 0; t < 16; ++t) acc[t] = 0.0f;
        for (int c = 0; c < 256; ++c) {
            float w = Wv[tid * 256 + c];
            for (int t = 0; t < 16; ++t) acc[t] += w * tok[c * 16 + t];
        }
        __syncthreads();
        for (int t = 0; t < 16; ++t) bufA[tid * 16 + t] = acc[t];
    }
    __syncthreads();
    // T_mid -> bufB: tm[c][t] = sum_s att[t][s]*vals[c][s]
    {
        float tm[16];
        for (int t = 0; t < 16; ++t) {
            float a = 0.0f;
            for (int s = 0; s < 16; ++s) a += att[t * 16 + s] * bufA[tid * 16 + s];
            tm[t] = a;
        }
        __syncthreads();
        for (int t = 0; t < 16; ++t) bufB[tid * 16 + t] = tm[t];
    }
    __syncthreads();
    // T_out -> bufA: tok + Wemb @ T_mid
    {
        float acc[16];
        for (int t = 0; t < 16; ++t) acc[t] = tok[tid * 16 + t];
        for (int c = 0; c < 256; ++c) {
            float w = Wemb[tid * 256 + c];
            for (int t = 0; t < 16; ++t) acc[t] += w * bufB[c * 16 + t];
        }
        __syncthreads();
        for (int t = 0; t < 16; ++t) bufA[tid * 16 + t] = acc[t];
    }
    __syncthreads();
    // T_P = Wt @ T_out -> global [b][c][t]
    {
        float acc[16];
        for (int t = 0; t < 16; ++t) acc[t] = 0.0f;
        for (int c = 0; c < 256; ++c) {
            float w = Wt[tid * 256 + c];
            for (int t = 0; t < 16; ++t) acc[t] += w * bufA[c * 16 + t];
        }
        for (int t = 0; t < 16; ++t) TP[(size_t)(b * 256 + tid) * 16 + t] = acc[t];
    }
}

// ---------------------------------------------------------------------------
// Token -> point: dm softmax + df.  16 points per block; reads xp (cols 256..511),
// writes df into cols 0..255 of the same scratch matrix (xq is dead by now).
// ---------------------------------------------------------------------------
__launch_bounds__(256)
__global__ void k_point(float* __restrict__ Ycat, const int* __restrict__ bid,
                        const float* __restrict__ TP) {
    __shared__ float xps[16 * 256];
    __shared__ float dms[16 * 16];
    __shared__ int   bp[16];
    int tid = threadIdx.x;
    int r0  = blockIdx.x * 16;

    for (int i = 0; i < 16; ++i) {
        int idx = tid + i * 256;
        int p = idx >> 8, c = idx & 255;
        xps[idx] = Ycat[(size_t)(r0 + p) * WIDE + 256 + c];
    }
    if (tid < 16) bp[tid] = bid[r0 + tid];
    __syncthreads();

    int p = tid >> 4, t = tid & 15;
    const float* tpb = TP + (size_t)bp[p] * 256 * 16;
    float a = 0.0f;
    for (int c = 0; c < 256; ++c) a += xps[p * 256 + c] * tpb[c * 16 + t];
    dms[p * 16 + t] = a;
    __syncthreads();

    float mx = -3.0e38f;
    for (int j = 0; j < 16; ++j) mx = fmaxf(mx, dms[p * 16 + j]);
    float e = expf(a - mx);
    __syncthreads();
    dms[p * 16 + t] = e;
    __syncthreads();
    float sum = 0.0f;
    for (int j = 0; j < 16; ++j) sum += dms[p * 16 + j];
    float w = e / sum;
    __syncthreads();
    dms[p * 16 + t] = w;
    __syncthreads();

    int cl = t;
    for (int i = 0; i < 16; ++i) {
        int c = i * 16 + cl;
        float acc = 0.0f;
        for (int tt = 0; tt < 16; ++tt) acc += dms[p * 16 + tt] * tpb[c * 16 + tt];
        Ycat[(size_t)(r0 + p) * WIDE + c] = acc;   // df
    }
}

// BatchNorm stats (deterministic two-stage)
__global__ void k_bnstat(const float* __restrict__ Ycat, float* __restrict__ psum,
                         float* __restrict__ psq) {
    int c = threadIdx.x;
    int n0 = blockIdx.x * 256;
    float s = 0.0f, q = 0.0f;
    for (int i = 0; i < 256; ++i) {
        float v = Ycat[(size_t)(n0 + i) * WIDE + 256 + c];   // xr
        s += v; q += v * v;
    }
    psum[blockIdx.x * 256 + c] = s;
    psq[blockIdx.x * 256 + c]  = q;
}

__global__ void k_bnred(const float* __restrict__ psum, const float* __restrict__ psq,
                        float* __restrict__ cs, float* __restrict__ cq) {
    int c = threadIdx.x;
    float s = 0.0f, q = 0.0f;
    for (int i = 0; i < NPTS / 256; ++i) { s += psum[i * 256 + c]; q += psq[i * 256 + c]; }
    cs[c] = s; cq[c] = q;
}

// Epilogue: BN(train) + ReLU + residual
__global__ void k_final(const float* __restrict__ xf, const float* __restrict__ Ycat,
                        const float* __restrict__ cs, const float* __restrict__ cq,
                        const float* __restrict__ gamma, const float* __restrict__ beta,
                        float* __restrict__ out) {
    int idx = blockIdx.x * 256 + threadIdx.x;
    int c = idx & 255;
    int n = idx >> 8;
    float mean = cs[c] * (1.0f / NPTS);
    float var  = cq[c] * (1.0f / NPTS) - mean * mean;
    float xr   = Ycat[(size_t)n * WIDE + 256 + c];
    float y    = gamma[c] * (xr - mean) * rsqrtf(var + EPSV) + beta[c];
    out[idx]   = xf[idx] + fmaxf(y, 0.0f);
}

// ---------------------------------------------------------------------------
extern "C" void kernel_launch(void* const* d_in, const int* in_sizes, int n_in,
                              void* d_out, int out_size, void* d_ws, size_t ws_size,
                              hipStream_t stream) {
    const float* x_f  = (const float*)d_in[0];
    const int*   bid  = (const int*)  d_in[1];
    const float* Wq   = (const float*)d_in[2];
    const float* Wk   = (const float*)d_in[3];
    const float* Wp   = (const float*)d_in[4];
    const float* Wv   = (const float*)d_in[5];
    const float* Wke  = (const float*)d_in[6];
    const float* Wqe  = (const float*)d_in[7];
    const float* Wemb = (const float*)d_in[8];
    const float* Wt   = (const float*)d_in[9];
    const float* Wtr  = (const float*)d_in[10];
    const float* gam  = (const float*)d_in[11];
    const float* bet  = (const float*)d_in[12];
    float* out = (float*)d_out;
    (void)in_sizes; (void)n_in; (void)out_size; (void)ws_size;

    char* wsp = (char*)d_ws;
    size_t off = 0;
    auto alloc = [&](size_t bytes) -> void* {
        void* p = wsp + off;
        off = (off + bytes + 255) & ~(size_t)255;
        return p;
    };
    float*  Ycat   = (float*) alloc((size_t)NPTS * WIDE * 4);      // xq|xp|xk -> df|xr|xk
    bf16_t* WcatF  = (bf16_t*)alloc((size_t)NCF_CAT * 8 * 512 * 2);
    bf16_t* WtbF   = (bf16_t*)alloc((size_t)NCF_TR * 8 * 512 * 2);
    float*  tokens = (float*) alloc((size_t)32768 * 4);
    float*  tpart  = (float*) alloc((size_t)32 * 32768 * 4);
    float*  TP     = (float*) alloc((size_t)32768 * 4);
    float*  mbuf   = (float*) alloc(128 * 4);
    float*  zbuf   = (float*) alloc(128 * 4);
    int*    start  = (int*)   alloc(16 * 4);
    float*  psum   = (float*) alloc((size_t)(NPTS / 256) * 256 * 4);
    float*  psq    = (float*) alloc((size_t)(NPTS / 256) * 256 * 4);
    float*  csum   = (float*) alloc(256 * 4);
    float*  csq    = (float*) alloc(256 * 4);

    k_convert<<<(NCF_CAT * 8 * 512 + NCF_TR * 8 * 512) / 256, 256, 0, stream>>>(
        Wq, Wk, Wp, Wtr, WcatF, WtbF);
    k_bounds<<<1, 32, 0, stream>>>(bid, start);
    // fused xq|xp|xk = x_f @ [Wq|Wp|Wk]
    k_gemm_bf16<<<dim3(NPTS / 128, WIDE / 64), 256, 0, stream>>>(x_f, 256, WcatF, Ycat, WIDE);
    k_segmax<<<BSEG * TTOK, 256, 0, stream>>>(Ycat, start, mbuf);
    k_segsum<<<BSEG * TTOK, 256, 0, stream>>>(Ycat, start, mbuf, zbuf);
    k_tokens<<<dim3(BSEG, 16, 32), 256, 0, stream>>>(Ycat, start, mbuf, zbuf, tpart);
    k_tokred<<<32768 / 256, 256, 0, stream>>>(tpart, tokens);
    k_tokattn<<<BSEG, 256, 0, stream>>>(tokens, Wv, Wke, Wqe, Wemb, Wt, TP);
    k_point<<<NPTS / 16, 256, 0, stream>>>(Ycat, bid, TP);
    // xr = df @ Wtrans  (df in cols 0..255, xr into cols 256..511)
    k_gemm_bf16<<<dim3(NPTS / 128, 256 / 64), 256, 0, stream>>>(Ycat, WIDE, WtbF, Ycat + 256, WIDE);
    k_bnstat<<<NPTS / 256, 256, 0, stream>>>(Ycat, psum, psq);
    k_bnred<<<1, 256, 0, stream>>>(psum, psq, csum, csq);
    k_final<<<(NPTS * CCH) / 256, 256, 0, stream>>>(x_f, Ycat, csum, csq, gam, bet, out);
}
